// ConvertedLlamaAttention_61847529062812
// MI455X (gfx1250) — compile-verified, matrix operand order checked by reading, not run on hardware
//
#include <hip/hip_runtime.h>

#define S_LEN 2048
#define HID 4096
#define NH 32
#define HD 128
#define D2 64
#define LR 8

typedef _Float16 v8h  __attribute__((ext_vector_type(8)));
typedef _Float16 v16h __attribute__((ext_vector_type(16)));
typedef float    v8f  __attribute__((ext_vector_type(8)));

// Load a 16x32 f16 A/B WMMA fragment row slice from a row-major buffer.
// rowptr points at (row, kstep_base). Lane layout per CDNA5 ISA:
//   element i<8  : k = 8*g + i
//   element i>=8 : k = 16 + 8*g + (i-8)
__device__ inline v16h frag_ld(const _Float16* rowptr, int g) {
  v8h lo = *(const v8h*)(rowptr + g * 8);
  v8h hi = *(const v8h*)(rowptr + 16 + g * 8);
  return __builtin_shufflevector(lo, hi, 0, 1, 2, 3, 4, 5, 6, 7,
                                 8, 9, 10, 11, 12, 13, 14, 15);
}

// ---------------------------------------------------------------------------
// RoPE cos/sin table: cos/sin of s * ROPE_BASE^(-i/64)
// ---------------------------------------------------------------------------
__global__ void rope_table_kernel(float* __restrict__ cosT, float* __restrict__ sinT) {
  int s = blockIdx.x;
  int i = threadIdx.x;  // 64 threads
  float inv = powf(10000.0f, -(float)i / 64.0f);
  float a = (float)s * inv;
  cosT[s * D2 + i] = cosf(a);
  sinT[s * D2 + i] = sinf(a);
}

// ---------------------------------------------------------------------------
// xa[s][r] = sum_k X[s,k] * A[r,k]  for the three LoRA A matrices (rank 8)
// grid: (2048, 3), block: 256
// ---------------------------------------------------------------------------
__global__ __launch_bounds__(256) void lora_xa_kernel(
    const float* __restrict__ X, const float* __restrict__ Aq,
    const float* __restrict__ Ak, const float* __restrict__ Av,
    float* __restrict__ xq, float* __restrict__ xk, float* __restrict__ xv) {
  int s = blockIdx.x;
  int p = blockIdx.y;
  int tid = threadIdx.x;
  const float* A = (p == 0) ? Aq : ((p == 1) ? Ak : Av);
  float* out = (p == 0) ? xq : ((p == 1) ? xk : xv);

  float part[LR];
#pragma unroll
  for (int r = 0; r < LR; ++r) part[r] = 0.0f;
  const float* xr = X + (size_t)s * HID;
  for (int k = tid; k < HID; k += 256) {
    float xv_ = xr[k];
#pragma unroll
    for (int r = 0; r < LR; ++r) part[r] += xv_ * A[(size_t)r * HID + k];
  }
  __shared__ float red[256];
  for (int r = 0; r < LR; ++r) {
    red[tid] = part[r];
    __syncthreads();
    for (int st = 128; st > 0; st >>= 1) {
      if (tid < st) red[tid] += red[tid + st];
      __syncthreads();
    }
    if (tid == 0) out[(size_t)s * LR + r] = red[0];
    __syncthreads();
  }
}

// ---------------------------------------------------------------------------
// Fused GEMM:  C = X(2048x4096) @ W(4096x4096)^T  with epilogue.
//   MODE 0/1 : + LoRA interleave + RoPE, f16 head-major out  (Q, K)
//   MODE 2   : + LoRA interleave,        f16 head-major out  (V)
//   MODE 3   : plain,                    f32 row-major out   (O-proj)
// XF16 selects f16 (attn) vs fp32 (hidden_states) activation input; W is
// fp32, converted to f16 while staging in LDS. 64-deep K slab per barrier.
// grid: (N/128=32, M/128=16), block 256 (8 waves: 4 M-waves x 2 N-waves).
// ---------------------------------------------------------------------------
template <int MODE, bool XF16>
__global__ __launch_bounds__(256) void gemm_fused_kernel(
    const float* __restrict__ Xf, const _Float16* __restrict__ Xh,
    const float* __restrict__ W, const float* __restrict__ xa,
    const float* __restrict__ Bm, const float* __restrict__ cosT,
    const float* __restrict__ sinT, _Float16* __restrict__ outh,
    float* __restrict__ outf) {
  __shared__ _Float16 Xs[128][72];   // 64-wide K slab, padded (stride 144B)
  __shared__ _Float16 Ws[128][72];
  __shared__ float    Cs[32][132];   // quarter-tile staging for epilogue

  const int tid = threadIdx.x;
  const int n0 = blockIdx.x * 128;
  const int m0 = blockIdx.y * 128;
  const int w = tid >> 5, lane = tid & 31;
  const int g = lane >> 4, r16 = lane & 15;
  const int wm = w >> 1;  // 0..3 -> 32-row band
  const int wn = w & 1;   // 0..1 -> 64-col band

  v8f acc[2][4] = {};

  const int lrow = tid >> 1;          // 0..127
  const int lcol = (tid & 1) * 32;    // 0 or 32

  for (int k0 = 0; k0 < HID; k0 += 64) {
    // prefetch next K slab of the streaming operands (global_prefetch_b8)
    if (k0 + 64 < HID) {
      __builtin_prefetch(W + (size_t)(n0 + lrow) * HID + k0 + 64 + lcol, 0, 1);
      if (XF16)
        __builtin_prefetch(Xh + (size_t)(m0 + lrow) * HID + k0 + 64 + lcol, 0, 1);
      else
        __builtin_prefetch(Xf + (size_t)(m0 + lrow) * HID + k0 + 64 + lcol, 0, 1);
    }
    // stage X tile (fp32->f16 or f16 passthrough)
    if (XF16) {
      const _Float16* px = Xh + (size_t)(m0 + lrow) * HID + k0 + lcol;
#pragma unroll
      for (int i = 0; i < 4; ++i)
        *(v8h*)&Xs[lrow][lcol + 8 * i] = ((const v8h*)px)[i];
    } else {
      const float* px = Xf + (size_t)(m0 + lrow) * HID + k0 + lcol;
#pragma unroll
      for (int i = 0; i < 8; ++i) {
        float4 t = ((const float4*)px)[i];
        Xs[lrow][lcol + 4 * i + 0] = (_Float16)t.x;
        Xs[lrow][lcol + 4 * i + 1] = (_Float16)t.y;
        Xs[lrow][lcol + 4 * i + 2] = (_Float16)t.z;
        Xs[lrow][lcol + 4 * i + 3] = (_Float16)t.w;
      }
    }
    // stage W tile (fp32->f16)
    {
      const float* pw = W + (size_t)(n0 + lrow) * HID + k0 + lcol;
#pragma unroll
      for (int i = 0; i < 8; ++i) {
        float4 t = ((const float4*)pw)[i];
        Ws[lrow][lcol + 4 * i + 0] = (_Float16)t.x;
        Ws[lrow][lcol + 4 * i + 1] = (_Float16)t.y;
        Ws[lrow][lcol + 4 * i + 2] = (_Float16)t.z;
        Ws[lrow][lcol + 4 * i + 3] = (_Float16)t.w;
      }
    }
    __syncthreads();

#pragma unroll
    for (int kk = 0; kk < 64; kk += 32) {
      v16h a[2], b[4];
#pragma unroll
      for (int mt = 0; mt < 2; ++mt)
        a[mt] = frag_ld(&Xs[wm * 32 + mt * 16 + r16][kk], g);
#pragma unroll
      for (int nt = 0; nt < 4; ++nt)
        b[nt] = frag_ld(&Ws[wn * 64 + nt * 16 + r16][kk], g);
#pragma unroll
      for (int mt = 0; mt < 2; ++mt)
#pragma unroll
        for (int nt = 0; nt < 4; ++nt)
          acc[mt][nt] = __builtin_amdgcn_wmma_f32_16x16x32_f16(
              false, a[mt], false, b[nt], (short)0, acc[mt][nt], false, false);
    }
    __syncthreads();
  }

  // Epilogue in four 32-row quarters (quarter q belongs to waves wm == q).
  for (int q4 = 0; q4 < 4; ++q4) {
    if (wm == q4) {
#pragma unroll
      for (int mt = 0; mt < 2; ++mt)
#pragma unroll
        for (int nt = 0; nt < 4; ++nt)
#pragma unroll
          for (int v = 0; v < 8; ++v)
            Cs[mt * 16 + v + 8 * g][wn * 64 + nt * 16 + r16] = acc[mt][nt][v];
    }
    __syncthreads();

    // LoRA interleave: out col j gets tmp col (j<2048 ? 2j : 2(j-2048)+1)
    if (MODE < 3) {
      for (int e = tid; e < 32 * 128; e += 256) {
        int ml = e >> 7, n = e & 127;
        int s = m0 + q4 * 32 + ml;
        int j = n0 + n;
        int c = (j < HID / 2) ? (2 * j) : (2 * (j - HID / 2) + 1);
        float t = 0.0f;
#pragma unroll
        for (int r = 0; r < LR; ++r)
          t += xa[(size_t)s * LR + r] * Bm[(size_t)c * LR + r];
        Cs[ml][n] += 2.0f * t;  // LORA_SCALING = 16/8
      }
      __syncthreads();
    }

    // RoPE (half-split) + store
    for (int e = tid; e < 32 * 128; e += 256) {
      int ml = e >> 7, d = e & 127;
      int s = m0 + q4 * 32 + ml;
      float val;
      if (MODE <= 1) {
        if (d < D2) {
          float c = cosT[(size_t)s * D2 + d], sn = sinT[(size_t)s * D2 + d];
          val = Cs[ml][d] * c - Cs[ml][d + D2] * sn;
        } else {
          int dd = d - D2;
          float c = cosT[(size_t)s * D2 + dd], sn = sinT[(size_t)s * D2 + dd];
          val = Cs[ml][dd] * sn + Cs[ml][d] * c;
        }
      } else {
        val = Cs[ml][d];
      }
      if (MODE < 3) {
        int h = n0 >> 7;  // N-tile == one head
        outh[(size_t)h * S_LEN * HD + (size_t)s * HD + d] = (_Float16)val;
      } else {
        outf[(size_t)s * HID + n0 + d] = val;
      }
    }
    __syncthreads();
  }
}

// ---------------------------------------------------------------------------
// Causal flash attention, one block per (q-tile of 128 rows, head).
// 8 waves, each owning 16 q rows. K processed in 64-key chunks.
// q/k/v are f16 head-major [h][s][128]; output attn is f16 [s][4096].
// ---------------------------------------------------------------------------
__global__ __launch_bounds__(256) void flash_attn_kernel(
    const _Float16* __restrict__ qh, const _Float16* __restrict__ kh,
    const _Float16* __restrict__ vh, _Float16* __restrict__ attn) {
  __shared__ _Float16 Ks[64][136];      // K chunk, row-major (n, d)
  __shared__ _Float16 Vs[128][72];      // V chunk transposed (d, j)
  __shared__ _Float16 Ps[8][16][72];    // per-wave P staging (m, j)

  const int tid = threadIdx.x;
  const int qt = blockIdx.x;
  const int h = blockIdx.y;
  const int w = tid >> 5, lane = tid & 31;
  const int g = lane >> 4, r16 = lane & 15;

  const _Float16* Q = qh + (size_t)h * S_LEN * HD;
  const _Float16* K = kh + (size_t)h * S_LEN * HD;
  const _Float16* V = vh + (size_t)h * S_LEN * HD;

  // Q fragments (16 rows x 128 d per wave), kept in registers
  v16h qf[4];
  {
    int s = qt * 128 + w * 16 + r16;
#pragma unroll
    for (int ks = 0; ks < 4; ++ks)
      qf[ks] = frag_ld(Q + (size_t)s * HD + 32 * ks, g);
  }

  float mrow[8], lrow[8];
  v8f o[8] = {};
#pragma unroll
  for (int v = 0; v < 8; ++v) {
    mrow[v] = -1e30f;
    lrow[v] = 0.0f;
  }

  const float scale = 0.08838834764831845f;  // 1/sqrt(128)
  const float LOG2E = 1.4426950408889634f;

  const int lr = tid >> 2;        // 0..63: row within chunk
  const int lq = (tid & 3) * 32;  // col base

  const int ktmax = 2 * qt + 1;
  for (int kt = 0; kt <= ktmax; ++kt) {
    // stage K chunk and transposed V chunk; prefetch next chunk
    {
      if (kt < ktmax) {
        __builtin_prefetch(K + (size_t)((kt + 1) * 64 + lr) * HD + lq, 0, 1);
        __builtin_prefetch(V + (size_t)((kt + 1) * 64 + lr) * HD + lq, 0, 1);
      }
      const _Float16* pk = K + (size_t)(kt * 64 + lr) * HD + lq;
#pragma unroll
      for (int i = 0; i < 4; ++i)
        *(v8h*)&Ks[lr][lq + 8 * i] = ((const v8h*)pk)[i];
      const _Float16* pv = V + (size_t)(kt * 64 + lr) * HD + lq;
#pragma unroll
      for (int i = 0; i < 4; ++i) {
        v8h t = ((const v8h*)pv)[i];
#pragma unroll
        for (int u = 0; u < 8; ++u) Vs[lq + 8 * i + u][lr] = t[u];
      }
    }
    __syncthreads();

    // scores: S = Q @ K^T  (16 x 64 per wave)
    v8f sc[4] = {};
#pragma unroll
    for (int ks = 0; ks < 4; ++ks)
#pragma unroll
      for (int nt = 0; nt < 4; ++nt) {
        v16h bk = frag_ld(&Ks[nt * 16 + r16][32 * ks], g);
        sc[nt] = __builtin_amdgcn_wmma_f32_16x16x32_f16(
            false, qf[ks], false, bk, (short)0, sc[nt], false, false);
      }

    // scale + causal mask + row max
    float pm[8];
#pragma unroll
    for (int v = 0; v < 8; ++v) pm[v] = -1e30f;
#pragma unroll
    for (int nt = 0; nt < 4; ++nt)
#pragma unroll
      for (int v = 0; v < 8; ++v) {
        int qidx = qt * 128 + w * 16 + v + 8 * g;
        int kidx = kt * 64 + nt * 16 + r16;
        float val = sc[nt][v] * scale;
        if (kidx > qidx) val = -1e30f;
        sc[nt][v] = val;
        pm[v] = fmaxf(pm[v], val);
      }
#pragma unroll
    for (int v = 0; v < 8; ++v)
#pragma unroll
      for (int off = 1; off < 16; off <<= 1)
        pm[v] = fmaxf(pm[v], __shfl_xor(pm[v], off, 32));

    // online softmax update
    float rs[8];
#pragma unroll
    for (int v = 0; v < 8; ++v) {
      float mn = fmaxf(mrow[v], pm[v]);
      float corr = exp2f((mrow[v] - mn) * LOG2E);
      mrow[v] = mn;
      lrow[v] *= corr;
      rs[v] = 0.0f;
#pragma unroll
      for (int dt = 0; dt < 8; ++dt) o[dt][v] *= corr;
    }
#pragma unroll
    for (int nt = 0; nt < 4; ++nt)
#pragma unroll
      for (int v = 0; v < 8; ++v) {
        float p = exp2f((sc[nt][v] - mrow[v]) * LOG2E);
        rs[v] += p;
        Ps[w][v + 8 * g][nt * 16 + r16] = (_Float16)p;
      }
#pragma unroll
    for (int v = 0; v < 8; ++v) {
#pragma unroll
      for (int off = 1; off < 16; off <<= 1) rs[v] += __shfl_xor(rs[v], off, 32);
      lrow[v] += rs[v];
    }

    // O += P @ V  (Ps is wave-private; in-wave DS ordering handled by waits)
#pragma unroll
    for (int ks2 = 0; ks2 < 2; ++ks2) {
      v16h ap = frag_ld(&Ps[w][r16][32 * ks2], g);
#pragma unroll
      for (int dt = 0; dt < 8; ++dt) {
        v16h bv = frag_ld(&Vs[dt * 16 + r16][32 * ks2], g);
        o[dt] = __builtin_amdgcn_wmma_f32_16x16x32_f16(
            false, ap, false, bv, (short)0, o[dt], false, false);
      }
    }
    __syncthreads();
  }

  // normalize + write f16 [s][4096]
#pragma unroll
  for (int dt = 0; dt < 8; ++dt)
#pragma unroll
    for (int v = 0; v < 8; ++v) {
      int s = qt * 128 + w * 16 + v + 8 * g;
      float val = o[dt][v] / lrow[v];
      attn[(size_t)s * HID + h * HD + dt * 16 + r16] = (_Float16)val;
    }
}

// ---------------------------------------------------------------------------
extern "C" void kernel_launch(void* const* d_in, const int* in_sizes, int n_in,
                              void* d_out, int out_size, void* d_ws,
                              size_t ws_size, hipStream_t stream) {
  const float* X  = (const float*)d_in[0];
  const float* Wq = (const float*)d_in[1];
  const float* Wk = (const float*)d_in[2];
  const float* Wv = (const float*)d_in[3];
  const float* Wo = (const float*)d_in[4];
  const float* Aq = (const float*)d_in[5];
  const float* Bq = (const float*)d_in[6];
  const float* Ak = (const float*)d_in[7];
  const float* Bk = (const float*)d_in[8];
  const float* Av = (const float*)d_in[9];
  const float* Bv = (const float*)d_in[10];

  char* ws = (char*)d_ws;
  const size_t MB = 1u << 20;
  _Float16* qh   = (_Float16*)(ws + 0 * MB);    // 16 MB each (f16 2048x4096)
  _Float16* kh   = (_Float16*)(ws + 16 * MB);
  _Float16* vh   = (_Float16*)(ws + 32 * MB);
  _Float16* attn = (_Float16*)(ws + 48 * MB);
  float* cosT = (float*)(ws + 64 * MB);         // 512 KB
  float* sinT = cosT + (size_t)S_LEN * D2;
  float* xaq  = sinT + (size_t)S_LEN * D2;      // 64 KB each
  float* xak  = xaq + (size_t)S_LEN * LR;
  float* xav  = xak + (size_t)S_LEN * LR;

  rope_table_kernel<<<S_LEN, D2, 0, stream>>>(cosT, sinT);
  lora_xa_kernel<<<dim3(S_LEN, 3), 256, 0, stream>>>(X, Aq, Ak, Av, xaq, xak, xav);

  dim3 gg(HID / 128, S_LEN / 128);  // (32, 16)
  gemm_fused_kernel<0, false><<<gg, 256, 0, stream>>>(
      X, nullptr, Wq, xaq, Bq, cosT, sinT, qh, nullptr);
  gemm_fused_kernel<1, false><<<gg, 256, 0, stream>>>(
      X, nullptr, Wk, xak, Bk, cosT, sinT, kh, nullptr);
  gemm_fused_kernel<2, false><<<gg, 256, 0, stream>>>(
      X, nullptr, Wv, xav, Bv, nullptr, nullptr, vh, nullptr);

  flash_attn_kernel<<<dim3(S_LEN / 128, NH), 256, 0, stream>>>(qh, kh, vh, attn);

  gemm_fused_kernel<3, true><<<gg, 256, 0, stream>>>(
      nullptr, attn, Wo, nullptr, nullptr, nullptr, nullptr, nullptr,
      (float*)d_out);
}
